// Classifier_37726992728702
// MI455X (gfx1250) — matrix operand order repaired and checked by reading, not compiled, not run on hardware
//
#include <hip/hip_runtime.h>
#include <hip/hip_bf16.h>

typedef __bf16 bf16_t;
typedef __attribute__((ext_vector_type(16))) __bf16 v16bf;
typedef __attribute__((ext_vector_type(8)))  __bf16 v8bf;
typedef __attribute__((ext_vector_type(8)))  float  v8f;

#define SEQ   256
#define BATCH 64
#define HID   256
#define EMB   256
#define OUTN  512
#define KDIM  512              // 2*HID
#define ENC_K 65536            // SEQ*EMB

static __device__ inline v8f zero8() {
    v8f z;
#pragma unroll
    for (int i = 0; i < 8; ++i) z[i] = 0.0f;
    return z;
}

static __device__ inline v8f wmma_bf16(v16bf a, v16bf b, v8f c) {
    return __builtin_amdgcn_wmma_f32_16x16x32_bf16(
        /*neg_a=*/false, a, /*neg_b=*/false, b,
        /*c_mod=*/(short)0, c, /*reuse_a=*/false, /*reuse_b=*/false);
}

static __device__ inline v16bf cat16(v8bf lo, v8bf hi) {
    return __builtin_shufflevector(lo, hi, 0, 1, 2, 3, 4, 5, 6, 7,
                                           8, 9, 10, 11, 12, 13, 14, 15);
}

// ---------------------------------------------------------------------------
// Kernel 1: convert the three step weights [512,256] f32 -> transposed bf16
// layout wT[w][n][k] so WMMA B operands are contiguous along K.
// ---------------------------------------------------------------------------
__global__ __launch_bounds__(256) void k_convert_weights(
    const float* __restrict__ i2h, const float* __restrict__ i2o,
    const float* __restrict__ o2o, bf16_t* __restrict__ wT) {
    const int total = 3 * KDIM * HID;
    for (int idx = blockIdx.x * blockDim.x + threadIdx.x; idx < total;
         idx += gridDim.x * blockDim.x) {
        const int w = idx / (KDIM * HID);
        const int r = idx % (KDIM * HID);
        const int k = r / HID;   // 0..511
        const int n = r % HID;   // 0..255
        const float* src = (w == 0) ? i2h : (w == 1) ? i2o : o2o;
        wT[(size_t)w * HID * KDIM + (size_t)n * KDIM + k] = (bf16_t)src[r];
    }
}

// ---------------------------------------------------------------------------
// Kernel 2: recurrent scan. One block = 16 batch rows, 8 waves.
// comb = [emb(x_t) | h_{t-1}] (bf16, LDS).  GEMM1: h,o.  GEMM2: out -> enc.
// ---------------------------------------------------------------------------
__global__ __launch_bounds__(256) void k_rnn(
    const int* __restrict__ x, const float* __restrict__ emb,
    const bf16_t* __restrict__ wT,
    const float* __restrict__ i2h_b, const float* __restrict__ i2o_b,
    const float* __restrict__ o2o_b, bf16_t* __restrict__ enc) {
    const bf16_t* wT_i2h = wT;
    const bf16_t* wT_i2o = wT + HID * KDIM;
    const bf16_t* wT_o2o = wT + 2 * HID * KDIM;

    __shared__ bf16_t comb[16][KDIM + 16];   // [inp | hidden]
    __shared__ bf16_t ho[16][KDIM + 16];     // [h | o]
    __shared__ int    toks[16];

    const int tid  = threadIdx.x;
    const int lane = tid & 31;
    const int wv   = tid >> 5;
    const int m0   = blockIdx.x * 16;

    const int arow = lane & 15;
    const int aoff = (lane & 16) ? 8 : 0;
    const int bn   = lane & 15;
    const int boff = (lane & 16) ? 8 : 0;
    const int mrow = (lane & 16) ? 8 : 0;

    // hidden_0 = 0
    for (int e = tid; e < 16 * HID; e += 256)
        comb[e >> 8][HID + (e & 255)] = (bf16_t)0.0f;
    __syncthreads();

    for (int t = 0; t < SEQ; ++t) {
        if (tid < 16) toks[tid] = x[t * BATCH + m0 + tid];
        __syncthreads();

        // gather embedding rows -> comb[:, 0:256] (f32 -> bf16)
        {
            const int m  = tid >> 4;
            const int c0 = (tid & 15) * 16;
            const float* er = emb + (size_t)toks[m] * HID;
#pragma unroll
            for (int j = 0; j < 16; ++j)
                comb[m][c0 + j] = (bf16_t)er[c0 + j];
        }
        __syncthreads();

        // ---- GEMM1: [16,512] x [512,256] twice (i2h, i2o); 4 tiles/wave ----
        v8f acc[4];
        const bf16_t* wsel[4];
        const float*  bsel[4];
        int nbase[4], ncol[4];
#pragma unroll
        for (int i = 0; i < 4; ++i) {
            const int tt   = wv * 4 + i;          // 0..31
            const bool iso = tt >= 16;
            const int n0   = (iso ? tt - 16 : tt) * 16;
            wsel[i]  = (iso ? wT_i2o : wT_i2h) + (size_t)n0 * KDIM;
            bsel[i]  = iso ? i2o_b : i2h_b;
            nbase[i] = n0;
            ncol[i]  = (iso ? HID : 0) + n0;
            acc[i]   = zero8();
        }
        for (int k0 = 0; k0 < KDIM; k0 += 32) {
            v8bf alo = *(const v8bf*)&comb[arow][k0 + aoff];
            v8bf ahi = *(const v8bf*)&comb[arow][k0 + aoff + 16];
            v16bf a  = cat16(alo, ahi);
#pragma unroll
            for (int i = 0; i < 4; ++i) {
                v16bf b = cat16(*(const v8bf*)(wsel[i] + (size_t)bn * KDIM + k0 + boff),
                                *(const v8bf*)(wsel[i] + (size_t)bn * KDIM + k0 + boff + 16));
                acc[i] = wmma_bf16(a, b, acc[i]);
            }
        }
#pragma unroll
        for (int i = 0; i < 4; ++i) {
            const float bias = bsel[i][nbase[i] + bn];
#pragma unroll
            for (int r = 0; r < 8; ++r)
                ho[r + mrow][ncol[i] + bn] = (bf16_t)(acc[i][r] + bias);
        }
        __syncthreads();

        // carry hidden: comb[:, 256:512] = h
        for (int e = tid; e < 16 * HID; e += 256)
            comb[e >> 8][HID + (e & 255)] = ho[e >> 8][e & 255];

        // ---- GEMM2: [16,512] x [512,256] (o2o); 2 tiles/wave -> enc ----
        v8f acc2[2];
        acc2[0] = zero8();
        acc2[1] = zero8();
        for (int k0 = 0; k0 < KDIM; k0 += 32) {
            v8bf alo = *(const v8bf*)&ho[arow][k0 + aoff];
            v8bf ahi = *(const v8bf*)&ho[arow][k0 + aoff + 16];
            v16bf a  = cat16(alo, ahi);
#pragma unroll
            for (int i = 0; i < 2; ++i) {
                const int n0 = (wv * 2 + i) * 16;
                v16bf b = cat16(*(const v8bf*)(wT_o2o + (size_t)(n0 + bn) * KDIM + k0 + boff),
                                *(const v8bf*)(wT_o2o + (size_t)(n0 + bn) * KDIM + k0 + boff + 16));
                acc2[i] = wmma_bf16(a, b, acc2[i]);
            }
        }
#pragma unroll
        for (int i = 0; i < 2; ++i) {
            const int n0 = (wv * 2 + i) * 16;
            const float bias = o2o_b[n0 + bn];
#pragma unroll
            for (int r = 0; r < 8; ++r)
                enc[(size_t)(m0 + r + mrow) * ENC_K + (size_t)t * EMB + n0 + bn] =
                    (bf16_t)(acc2[i][r] + bias);
        }
        __syncthreads();
    }
}

// ---------------------------------------------------------------------------
// Kernel 3: logits = enc[64,65536](bf16) @ pred_w[65536,512](f32->bf16 in regs)
// One block per 16-wide N strip (pred_w read exactly once from HBM).
// 16 waves: wave w -> (m_tile = w&3, k_part = w>>2), LDS reduce over k parts.
// ---------------------------------------------------------------------------
__global__ __launch_bounds__(512) void k_pred_gemm(
    const bf16_t* __restrict__ enc, const float* __restrict__ pred_w,
    const float* __restrict__ pred_b, float* __restrict__ logits) {
    const int tid  = threadIdx.x;
    const int lane = tid & 31;
    const int wv   = tid >> 5;
    const int n0   = blockIdx.x * 16;

    const int m_tile = wv & 3;
    const int k_part = wv >> 2;                  // 0..3, each 16384 of K
    const int arow   = m_tile * 16 + (lane & 15);
    const int aoff   = (lane & 16) ? 8 : 0;
    const int bn     = n0 + (lane & 15);
    const int boff   = (lane & 16) ? 8 : 0;

    v8f acc = zero8();
    const int kbase = k_part * (ENC_K / 4);
    const bf16_t* arow_p = enc + (size_t)arow * ENC_K;
    for (int kk = 0; kk < ENC_K / 4; kk += 32) {
        const int k = kbase + kk;
        v8bf alo = *(const v8bf*)(arow_p + k + aoff);
        v8bf ahi = *(const v8bf*)(arow_p + k + aoff + 16);
        v16bf a  = cat16(alo, ahi);
        v16bf b;
#pragma unroll
        for (int j = 0; j < 16; ++j)
            b[j] = (bf16_t)pred_w[(size_t)(k + boff + (j & 7) + ((j & 8) ? 16 : 0)) * OUTN + bn];
        acc = wmma_bf16(a, b, acc);
    }

    __shared__ float red[16][16][17];            // [wave][m][n] (+pad)
#pragma unroll
    for (int r = 0; r < 8; ++r)
        red[wv][r + ((lane & 16) ? 8 : 0)][lane & 15] = acc[r];
    __syncthreads();

    for (int e = tid; e < 1024; e += 512) {
        const int mt = e >> 8;
        const int m  = (e >> 4) & 15;
        const int nn = e & 15;
        const float v = red[mt][m][nn] + red[mt + 4][m][nn] +
                        red[mt + 8][m][nn] + red[mt + 12][m][nn] +
                        pred_b[n0 + nn];
        logits[(size_t)(mt * 16 + m) * OUTN + n0 + nn] = v;
    }
}

// ---------------------------------------------------------------------------
// Kernel 4: log_softmax + argmax + exp per batch row.
// d_out layout: preds[64] | logprobs[64*512] | probs[64*512]  (all f32)
// ---------------------------------------------------------------------------
__global__ __launch_bounds__(512) void k_softmax(
    const float* __restrict__ logits, float* __restrict__ out) {
    const int m = blockIdx.x, tid = threadIdx.x;
    __shared__ float sv[512];
    __shared__ int   si[512];

    const float v = logits[(size_t)m * OUTN + tid];
    sv[tid] = v;
    si[tid] = tid;
    __syncthreads();
    for (int s = 256; s > 0; s >>= 1) {
        if (tid < s) {
            const float a = sv[tid], b = sv[tid + s];
            const int ia = si[tid], ib = si[tid + s];
            if (b > a || (b == a && ib < ia)) { sv[tid] = b; si[tid] = ib; }
        }
        __syncthreads();
    }
    const float vmax = sv[0];
    const int   imax = si[0];
    __syncthreads();

    sv[tid] = expf(v - vmax);
    __syncthreads();
    for (int s = 256; s > 0; s >>= 1) {
        if (tid < s) sv[tid] += sv[tid + s];
        __syncthreads();
    }
    const float lse = vmax + logf(sv[0]);
    const float lp  = v - lse;
    out[BATCH + (size_t)m * OUTN + tid] = lp;
    out[BATCH + BATCH * OUTN + (size_t)m * OUTN + tid] = expf(lp);
    if (tid == 0) out[m] = (float)imax;
}

// ---------------------------------------------------------------------------
extern "C" void kernel_launch(void* const* d_in, const int* in_sizes, int n_in,
                              void* d_out, int out_size, void* d_ws,
                              size_t ws_size, hipStream_t stream) {
    const int*   x      = (const int*)d_in[0];
    const float* emb    = (const float*)d_in[1];
    const float* i2h_w  = (const float*)d_in[2];
    const float* i2h_b  = (const float*)d_in[3];
    const float* i2o_w  = (const float*)d_in[4];
    const float* i2o_b  = (const float*)d_in[5];
    const float* o2o_w  = (const float*)d_in[6];
    const float* o2o_b  = (const float*)d_in[7];
    const float* pred_w = (const float*)d_in[8];
    const float* pred_b = (const float*)d_in[9];
    float* out = (float*)d_out;

    char* ws = (char*)d_ws;
    bf16_t* wT     = (bf16_t*)ws;                             // 786,432 B
    bf16_t* enc    = (bf16_t*)(ws + 786432);                  // 8,388,608 B
    float*  logits = (float*)(ws + 786432 + 8388608);         // 131,072 B

    k_convert_weights<<<192, 256, 0, stream>>>(i2h_w, i2o_w, o2o_w, wT);
    k_rnn<<<4, 256, 0, stream>>>(x, emb, wT, i2h_b, i2o_b, o2o_b, enc);
    k_pred_gemm<<<32, 512, 0, stream>>>(enc, pred_w, pred_b, logits);
    k_softmax<<<64, 512, 0, stream>>>(logits, out);
}
